// QVLoraExpertRouter_89498528514129
// MI455X (gfx1250) — matrix operand order repaired and checked by reading, not compiled, not run on hardware
//
#include <hip/hip_runtime.h>
#include <hip/hip_bf16.h>

// ---------------------------------------------------------------------------
// QVLoraExpertRouter for MI455X (gfx1250, wave32, WMMA)
//   T=16384 tokens, D=2048, E=8, R=16, QO=VO=2048
//   Pipeline:
//     prep_weights : fp32 -> bf16, N-major transposed layouts for WMMA B-frags
//     router       : wave-per-token logits -> softmax -> top2 -> dense w[T,8]*2
//     gemm1_low    : [T,2048] x [2048,256] bf16 WMMA, scale by w, -> low q/v bf16
//     gemm2_delta  : [T,128] x [128,2048] bf16 WMMA -> fp32 outputs (x2: q,v)
// ---------------------------------------------------------------------------

typedef __attribute__((ext_vector_type(16))) __bf16 v16bf;
typedef __attribute__((ext_vector_type(8)))  float  v8f;
typedef int v4i __attribute__((vector_size(16)));
typedef __attribute__((address_space(3))) v4i* lds_v4i_p;

#define T_TOK 16384
#define DDIM  2048
#define NE    8
#define RNK   16
#define NOUT  2048
#define KLOW  128   // E*R joint contraction dim for gemm2

#if __has_builtin(__builtin_amdgcn_global_load_async_to_lds_b128)
#define HAVE_ASYNC_LDS 1
#else
#define HAVE_ASYNC_LDS 0
#endif

__device__ __forceinline__ void wait_asynccnt0() {
#if __has_builtin(__builtin_amdgcn_s_wait_asynccnt)
  __builtin_amdgcn_s_wait_asynccnt(0);
#else
  asm volatile("s_wait_asynccnt 0x0" ::: "memory");
#endif
}

// ---- WMMA helpers (layouts per CDNA5 ISA §7.12.2) ---------------

__device__ __forceinline__ v8f wmma_bf16(v16bf a, v16bf b, v8f c) {
  // 8 args: (neg_a, A, neg_b, B, c_mod, C, reuse_a, reuse_b)
  return __builtin_amdgcn_wmma_f32_16x16x32_bf16(false, a, false, b, (short)0, c,
                                                 false, false);
}

// A fragment: 16x32 bf16 tile, row-major in LDS with given element stride.
// Lane l (l<16): row=l, elements = K[0..7] then K[16..23]; lanes 16-31: +8.
__device__ __forceinline__ v16bf load_a_frag(const __bf16* base, int stride) {
  int lane = threadIdx.x & 31;
  const __bf16* p = base + (lane & 15) * stride + ((lane & 16) ? 8 : 0);
  union { v16bf v; uint4 q[2]; } u;
  u.q[0] = *(const uint4*)(p);       // K = kb .. kb+7
  u.q[1] = *(const uint4*)(p + 16);  // K = kb+16 .. kb+23
  return u.v;
}

// B fragment: 32x16 bf16 tile from an N-major (transposed) weight matrix
// bt[N][ldk]. Lane l (l<16): col=l, K = k0..k0+7 then k0+16..k0+23; lanes 16-31: +8.
// Two 8-bf16 runs per lane -> two 16B global loads.
__device__ __forceinline__ v16bf load_b_frag(const __bf16* __restrict__ bt,
                                             int ldk, int nbase, int k0) {
  int lane = threadIdx.x & 31;
  const __bf16* p = bt + (size_t)(nbase + (lane & 15)) * ldk + k0 +
                    ((lane & 16) ? 8 : 0);
  union { v16bf v; uint4 q[2]; } u;
  u.q[0] = *(const uint4*)(p);
  u.q[1] = *(const uint4*)(p + 16);
  return u.v;
}

// ---- weight prep: build bf16 transposed operands --------------------------
// Acat_t[256][2048]: row n = (q: e*16+r | v: 128+e*16+r), col d.  (B of gemm1)
// Bq_t/Bv_t[2048][128]: row n (output col), col k=e*16+r.        (B of gemm2)
__global__ __launch_bounds__(256) void prep_weights(
    const float* __restrict__ qa, const float* __restrict__ qb,
    const float* __restrict__ va, const float* __restrict__ vb,
    __bf16* __restrict__ Acat_t, __bf16* __restrict__ Bqt,
    __bf16* __restrict__ Bvt) {
  int idx = blockIdx.x * 256 + threadIdx.x;
  if (idx < 256 * DDIM) {                       // Acat_t
    int n = idx >> 11, d = idx & (DDIM - 1);
    int e = (n & 127) >> 4, r = n & 15;
    const float* src = (n < 128) ? qa : va;
    Acat_t[idx] = (__bf16)src[((size_t)e * DDIM + d) * RNK + r];
  } else if (idx < 256 * DDIM + NOUT * KLOW) {  // Bq_t
    int j = idx - 256 * DDIM;
    int n = j >> 7, k = j & 127;
    int e = k >> 4, r = k & 15;
    Bqt[j] = (__bf16)qb[((size_t)e * RNK + r) * NOUT + n];
  } else {                                      // Bv_t
    int j = idx - 256 * DDIM - NOUT * KLOW;
    int n = j >> 7, k = j & 127;
    int e = k >> 4, r = k & 15;
    Bvt[j] = (__bf16)vb[((size_t)e * RNK + r) * NOUT + n];
  }
}

// ---- router: one wave32 per token -----------------------------------------
__global__ __launch_bounds__(256) void router_kernel(
    const float* __restrict__ h, const float* __restrict__ rw,
    float* __restrict__ wdense) {
  int wave = threadIdx.x >> 5, lane = threadIdx.x & 31;
  int t = blockIdx.x * 8 + wave;
  const float* hrow = h + (size_t)t * DDIM;
  float acc[NE];
#pragma unroll
  for (int e = 0; e < NE; ++e) acc[e] = 0.f;
  for (int d = lane; d < DDIM; d += 32) {
    float hv = hrow[d];
#pragma unroll
    for (int e = 0; e < NE; ++e) acc[e] += hv * rw[e * DDIM + d];
  }
#pragma unroll
  for (int e = 0; e < NE; ++e)
#pragma unroll
    for (int off = 16; off > 0; off >>= 1)
      acc[e] += __shfl_xor(acc[e], off, 32);
  // softmax over 8 logits (redundant on all lanes)
  float m = acc[0];
#pragma unroll
  for (int e = 1; e < NE; ++e) m = fmaxf(m, acc[e]);
  float p[NE], ps = 0.f;
#pragma unroll
  for (int e = 0; e < NE; ++e) { p[e] = __expf(acc[e] - m); ps += p[e]; }
  // top-2 (lowest index wins ties, like lax.top_k)
  int i1 = 0;
#pragma unroll
  for (int e = 1; e < NE; ++e) if (p[e] > p[i1]) i1 = e;
  int i2 = (i1 == 0) ? 1 : 0;
#pragma unroll
  for (int e = 0; e < NE; ++e) if (e != i1 && p[e] > p[i2]) i2 = e;
  float s1 = p[i1] / ps, s2 = p[i2] / ps;
  float inv = 2.0f / (s1 + s2 + 1e-20f);  // renorm * SCALE (alpha/rank = 2)
  if (lane < NE) {
    float we = (lane == i1) ? s1 * inv : (lane == i2) ? s2 * inv : 0.f;
    wdense[(size_t)t * NE + lane] = we;
  }
}

// ---- gemm1: low = (h x Acat) * w  -> bf16 [T,128] q and v -----------------
// Block: 256 thr = 8 waves; 16 rows x 256 cols; wave w -> col tiles 32w, 32w+16.
// A tile (16x32 fp32 -> bf16) staged per K-chunk: 512 elems = 1 float2/thread,
// stored as one packed 2xbf16 dword (branch-free, coalesced).
__global__ __launch_bounds__(256) void gemm1_low(
    const float* __restrict__ h, const __bf16* __restrict__ Acat_t,
    const float* __restrict__ wdense, __bf16* __restrict__ lowq,
    __bf16* __restrict__ lowv) {
  __shared__ __attribute__((aligned(16))) __bf16 Atile[16 * 32];
  int tid = threadIdx.x, wave = tid >> 5, lane = tid & 31;
  int r0 = blockIdx.x * 16;
  int n0 = wave * 32, n1 = n0 + 16;
  int srow = tid >> 4;              // 0..15
  int scol = (tid & 15) * 2;        // 0,2,...,30
  const float* hsrc = h + (size_t)(r0 + srow) * DDIM + scol;
  v8f acc0 = {}, acc1 = {};
  for (int k0 = 0; k0 < DDIM; k0 += 32) {
    __syncthreads();  // WAR: previous iter's frag reads done
    float2 hv = *(const float2*)(hsrc + k0);
    union { __bf16 b[2]; unsigned int u; } pk;
    pk.b[0] = (__bf16)hv.x;
    pk.b[1] = (__bf16)hv.y;
    *(unsigned int*)&Atile[srow * 32 + scol] = pk.u;
    if (k0 + 32 < DDIM) __builtin_prefetch(hsrc + k0 + 32, 0, 1);
    __syncthreads();
    v16bf a  = load_a_frag(Atile, 32);
    v16bf b0 = load_b_frag(Acat_t, DDIM, n0, k0);
    v16bf b1 = load_b_frag(Acat_t, DDIM, n1, k0);
    acc0 = wmma_bf16(a, b0, acc0);
    acc1 = wmma_bf16(a, b1, acc1);
  }
  int mb = (lane & 16) ? 8 : 0;
  int col0 = n0 + (lane & 15), col1 = n1 + (lane & 15);
  __bf16* base0 = (col0 < KLOW) ? lowq : lowv;
  __bf16* base1 = (col1 < KLOW) ? lowq : lowv;
  int cc0 = col0 & (KLOW - 1), cc1 = col1 & (KLOW - 1);
  int ex0 = cc0 >> 4, ex1 = cc1 >> 4;
#pragma unroll
  for (int r = 0; r < 8; ++r) {
    int row = r0 + mb + r;
    float w0 = wdense[(size_t)row * NE + ex0];
    float w1 = wdense[(size_t)row * NE + ex1];
    base0[(size_t)row * KLOW + cc0] = (__bf16)(acc0[r] * w0);
    base1[(size_t)row * KLOW + cc1] = (__bf16)(acc1[r] * w1);
  }
}

// ---- gemm2: delta = low x B^T  -> fp32 [T,2048] ---------------------------
// Block: 256 thr = 8 waves; 128 rows x 32 cols; full K=128 of A staged in LDS
// via async global->LDS B128 copies (ASYNCcnt path) when available.
__global__ __launch_bounds__(256) void gemm2_delta(
    const __bf16* __restrict__ low, const __bf16* __restrict__ Bt,
    float* __restrict__ out) {
  __shared__ __attribute__((aligned(16))) __bf16 Atile[128 * KLOW];  // 32 KB
  int tid = threadIdx.x, wave = tid >> 5, lane = tid & 31;
  int r0 = blockIdx.x * 128;
  int c0 = blockIdx.y * 32;
  // cooperative A-block copy: 128x128 bf16 = 2048 x 16B chunks
  const uint4* src = (const uint4*)(low + (size_t)r0 * KLOW);
  uint4* dst = (uint4*)Atile;
#if HAVE_ASYNC_LDS
#pragma unroll
  for (int ch = tid; ch < 2048; ch += 256) {
    __builtin_amdgcn_global_load_async_to_lds_b128(
        (v4i*)(src + ch), (lds_v4i_p)(dst + ch), 0, 0);
  }
  wait_asynccnt0();
#else
#pragma unroll
  for (int ch = tid; ch < 2048; ch += 256) dst[ch] = src[ch];
#endif
  __syncthreads();
  v8f acc0 = {}, acc1 = {};
  int rowt = wave * 16;
#pragma unroll
  for (int k0 = 0; k0 < KLOW; k0 += 32) {
    v16bf a  = load_a_frag(Atile + rowt * KLOW + k0, KLOW);
    v16bf b0 = load_b_frag(Bt, KLOW, c0, k0);
    v16bf b1 = load_b_frag(Bt, KLOW, c0 + 16, k0);
    acc0 = wmma_bf16(a, b0, acc0);
    acc1 = wmma_bf16(a, b1, acc1);
  }
  int mb = (lane & 16) ? 8 : 0;
  int crow = r0 + rowt + mb;
  int ccol = c0 + (lane & 15);
#pragma unroll
  for (int r = 0; r < 8; ++r) {
    out[(size_t)(crow + r) * NOUT + ccol]      = acc0[r];
    out[(size_t)(crow + r) * NOUT + ccol + 16] = acc1[r];
  }
}

// ---------------------------------------------------------------------------
extern "C" void kernel_launch(void* const* d_in, const int* in_sizes, int n_in,
                              void* d_out, int out_size, void* d_ws,
                              size_t ws_size, hipStream_t stream) {
  (void)in_sizes; (void)n_in; (void)out_size; (void)ws_size;
  const float* h  = (const float*)d_in[0];
  const float* rw = (const float*)d_in[1];
  const float* qa = (const float*)d_in[2];
  const float* qb = (const float*)d_in[3];
  const float* va = (const float*)d_in[4];
  const float* vb = (const float*)d_in[5];

  char* ws = (char*)d_ws;
  float*  wdense = (float*)(ws + 0);                      // 512 KB
  __bf16* Acat   = (__bf16*)(ws + 524288);                // 1 MB
  __bf16* Bqt    = (__bf16*)(ws + 1572864);               // 512 KB
  __bf16* Bvt    = (__bf16*)(ws + 2097152);               // 512 KB
  __bf16* lowq   = (__bf16*)(ws + 2621440);               // 4 MB
  __bf16* lowv   = (__bf16*)(ws + 6815744);               // 4 MB -> 10.5 MB total

  // 1) weight conversion/transposition: 1,048,576 elements
  prep_weights<<<4096, 256, 0, stream>>>(qa, qb, va, vb, Acat, Bqt, Bvt);
  // 2) router: 8 tokens per block
  router_kernel<<<T_TOK / 8, 256, 0, stream>>>(h, rw, wdense);
  // 3) low projection + scaling
  gemm1_low<<<T_TOK / 16, 256, 0, stream>>>(h, Acat, wdense, lowq, lowv);
  // 4) deltas (q then v), outputs concatenated in d_out
  dim3 g2(T_TOK / 128, NOUT / 32);
  float* outq = (float*)d_out;
  float* outv = outq + (size_t)T_TOK * NOUT;
  gemm2_delta<<<g2, 256, 0, stream>>>(lowq, Bqt, outq);
  gemm2_delta<<<g2, 256, 0, stream>>>(lowv, Bvt, outv);
}
